// MultiHead_Attention_46162308497789
// MI455X (gfx1250) — matrix operand order repaired and checked by reading, not compiled, not run on hardware
//
#include <hip/hip_runtime.h>
#include <cstdint>
#include <cstddef>

// Problem constants (match reference)
#define B_    4
#define S_    2048
#define HID_  1024
#define NH_   16
#define DH_   64
#define M_    (B_ * S_)          // 8192 rows for all GEMMs

typedef unsigned short u16;
typedef unsigned int   u32;
typedef unsigned long long u64;
typedef __bf16 bf16;
typedef bf16  v16bf __attribute__((ext_vector_type(16)));
typedef u16   v16us __attribute__((ext_vector_type(16)));
typedef float v8f   __attribute__((ext_vector_type(8)));
typedef u32   u32x4 __attribute__((ext_vector_type(4)));
typedef int   i32x4 __attribute__((ext_vector_type(4)));
typedef int   i32x8 __attribute__((ext_vector_type(8)));

#define AS3 __attribute__((address_space(3)))
#define AS1 __attribute__((address_space(1)))

// ---- CDNA5 feature probes (compile-safe fallbacks) -------------------------
#if __has_builtin(__builtin_amdgcn_tensor_load_to_lds)
#define HAVE_TDM 1
#else
#define HAVE_TDM 0
#endif

#if __has_builtin(__builtin_amdgcn_global_load_async_to_lds_b128)
#define HAVE_ASYNC 1
#else
#define HAVE_ASYNC 0
#endif

__device__ __forceinline__ void wait_tensor0() {
#if __has_builtin(__builtin_amdgcn_s_wait_tensorcnt)
    __builtin_amdgcn_s_wait_tensorcnt(0);
#else
    asm volatile("s_wait_tensorcnt 0" ::: "memory");
#endif
}

__device__ __forceinline__ void wait_async0() {
#if __has_builtin(__builtin_amdgcn_s_wait_asynccnt)
    __builtin_amdgcn_s_wait_asynccnt(0);
#else
    asm volatile("s_wait_asynccnt 0" ::: "memory");
#endif
}

struct U32x8 { uint4 lo, hi; };

__device__ __forceinline__ u16 f2bf(float f) {
    u32 u = __builtin_bit_cast(u32, f);
    u += 0x7fffu + ((u >> 16) & 1u);   // round-to-nearest-even
    return (u16)(u >> 16);
}

__device__ __forceinline__ v16bf mkfrag(uint4 lo, uint4 hi) {
    U32x8 t{lo, hi};
    return __builtin_bit_cast(v16bf, t);
}

__device__ __forceinline__ v8f wmma_bf16(v16bf a, v16bf b, v8f c) {
    return __builtin_amdgcn_wmma_f32_16x16x32_bf16(
        /*neg_a=*/false, a, /*neg_b=*/false, b,
        /*c_mod=*/(short)0, c, /*reuse_a=*/false, /*reuse_b=*/false);
}

#if HAVE_TDM
// Issue a TDM 2D tile load: tile_dim0=32 (k, contiguous), tile_dim1=64 (rows),
// row stride = Kdim elements, bf16 (data_size code 1 = 2 bytes).
// D# bit layout per CDNA5 ISA ch.8 (groups 0/1; groups 2/3 zero for 2-D).
__device__ __forceinline__ void tdm_issue_w(const u16* W, int Kdim, int Ndim,
                                            int n0, int kc, u32 lds_addr) {
    u64 ga = (u64)(uintptr_t)W +
             2ull * ((u64)(u32)n0 * (u32)Kdim + (u32)kc);
    u32x4 g0 = { 1u,                                   // count=1 (valid, user mode)
                 lds_addr,                             // lds_addr [63:32]
                 (u32)ga,                              // global_addr[31:0]
                 ((u32)(ga >> 32) & 0x01ffffffu) | (2u << 30) };  // ga[56:32] | type=2
    i32x8 g1 = { (int)0x00010000u,                             // data_size=1 (2B)
                 (int)(((u32)Kdim & 0xffffu) << 16),           // tensor_dim0 lo16
                 (int)(((u32)Kdim >> 16) | (((u32)Ndim & 0xffffu) << 16)), // dim0 hi | dim1 lo
                 (int)(((u32)Ndim >> 16) | (32u << 16)),       // dim1 hi | tile_dim0=32
                 (int)64,                                      // tile_dim1=64, tile_dim2=0
                 (int)(u32)Kdim,                               // tensor_dim0_stride lo32
                 0, 0 };
    i32x4 z4 = { 0, 0, 0, 0 };
#if __clang_major__ >= 23
    i32x8 z8 = { 0, 0, 0, 0, 0, 0, 0, 0 };
    __builtin_amdgcn_tensor_load_to_lds(g0, g1, z4, z4, z8, 0);
#else
    __builtin_amdgcn_tensor_load_to_lds(g0, g1, z4, z4, 0);
#endif
}
#endif  // HAVE_TDM

// ---------------------------------------------------------------------------
// fp32 -> bf16 conversion (vectorized, n divisible by 4)
// ---------------------------------------------------------------------------
__global__ __launch_bounds__(256)
void f32_to_bf16_kernel(const float* __restrict__ src, u16* __restrict__ dst, int n) {
    int i = (blockIdx.x * blockDim.x + threadIdx.x) * 4;
    if (i < n) {
        float4 f = *(const float4*)(src + i);
        ushort4 o;
        o.x = f2bf(f.x); o.y = f2bf(f.y); o.z = f2bf(f.z); o.w = f2bf(f.w);
        *(ushort4*)(dst + i) = o;
    }
}

// ---------------------------------------------------------------------------
// WMMA GEMM: C[m,n] = sum_k A[m,k] * W[n,k] + bias[n]   (nn.Linear semantics)
// A: [Mdim, Kdim] bf16 row-major, W: [Ndim, Kdim] bf16 row-major.
// mode 0: store fp32 row-major [Mdim, Ndim] to outf.
// mode 1: store bf16, permuted [B, NH, S, DH] (m=b*S+s, n=h*DH+d), scaled.
// Each wave computes a 16(M) x 64(N) tile; 4 waves/block cover 64 M-rows and
// share one 64-col W strip. The W tile (64x32 bf16 = 4KB) is staged per block
// via the Tensor Data Mover into double-buffered LDS: wave 0 issues the DMA
// for iteration i+1 while all waves run WMMAs on iteration i.
// ---------------------------------------------------------------------------
__global__ __launch_bounds__(128)
void linear_wmma_kernel(const u16* __restrict__ A, const u16* __restrict__ W,
                        const float* __restrict__ bias,
                        float* __restrict__ outf, u16* __restrict__ outbf,
                        int Mdim, int Ndim, int Kdim, float scale, int mode) {
    const int lane = threadIdx.x & 31;
    const int wv   = threadIdx.x >> 5;
    const int col  = lane & 15;     // N / M-row index inside fragments
    const int hi16 = lane >> 4;     // 0: lanes 0-15, 1: lanes 16-31

    const int m0 = (blockIdx.y * 4 + wv) * 16;
    const int n0 = blockIdx.x * 64;

    v8f zero = {0.f,0.f,0.f,0.f,0.f,0.f,0.f,0.f};
    v8f acc[4] = {zero, zero, zero, zero};

    const u16* arow = A + (size_t)(m0 + col) * Kdim;

#if HAVE_TDM
    __shared__ __align__(16) u16 wbuf[2][64][32];   // double-buffered W tile (8 KB)
    const u32 lds_base = (u32)(uintptr_t)(AS3 void*)&wbuf[0][0][0];

    if (wv == 0) tdm_issue_w(W, Kdim, Ndim, n0, 0, lds_base);

    for (int kc = 0; kc < Kdim; kc += 32) {
        const int cur = (kc >> 5) & 1;
        if (wv == 0) wait_tensor0();          // tile `kc` landed in wbuf[cur]
        __syncthreads();                      // visible to all waves
        if (wv == 0 && kc + 32 < Kdim)        // prefetch next tile under WMMAs
            tdm_issue_w(W, Kdim, Ndim, n0, kc + 32, lds_base + (u32)(cur ^ 1) * 4096u);

        __builtin_prefetch(arow + kc + 256, 0, 0);
        // A fragment, 16x32: lanes 0-15 hold K {0-7,16-23}, lanes 16-31 {8-15,24-31}
        uint4 alo = *(const uint4*)(arow + kc + hi16 * 8);
        uint4 ahi = *(const uint4*)(arow + kc + 16 + hi16 * 8);
        v16bf af = mkfrag(alo, ahi);
#pragma unroll
        for (int g = 0; g < 4; ++g) {
            // B fragment from LDS: lane holds col n=g*16+col, K {hi16*8+0..7, 16+hi16*8+0..7}
            const uint4* wp = (const uint4*)&wbuf[cur][g * 16 + col][hi16 * 8];
            acc[g] = wmma_bf16(af, mkfrag(wp[0], wp[2]), acc[g]);
        }
    }
#else
    for (int kc = 0; kc < Kdim; kc += 32) {
        __builtin_prefetch(arow + kc + 256, 0, 0);
        uint4 alo = *(const uint4*)(arow + kc + hi16 * 8);
        uint4 ahi = *(const uint4*)(arow + kc + 16 + hi16 * 8);
        v16bf af = mkfrag(alo, ahi);
#pragma unroll
        for (int g = 0; g < 4; ++g) {
            const u16* wrow = W + (size_t)(n0 + g * 16 + col) * Kdim + kc + hi16 * 8;
            uint4 blo = *(const uint4*)(wrow);
            uint4 bhi = *(const uint4*)(wrow + 16);
            acc[g] = wmma_bf16(af, mkfrag(blo, bhi), acc[g]);
        }
    }
#endif

#pragma unroll
    for (int g = 0; g < 4; ++g) {
        const int n = n0 + g * 16 + col;
        const float bn = bias[n];
#pragma unroll
        for (int r = 0; r < 8; ++r) {
            const int m = m0 + r + 8 * hi16;   // C layout: M = r + 8*(lane>=16)
            const float v = acc[g][r] + bn;
            if (mode == 0) {
                outf[(size_t)m * Ndim + n] = v;
            } else {
                const int bidx = m / S_, s = m % S_;
                const int hh = n / DH_, d = n % DH_;
                outbf[(((size_t)(bidx * NH_ + hh)) * S_ + s) * DH_ + d] = f2bf(v * scale);
            }
        }
    }
}

// ---------------------------------------------------------------------------
// Flash attention over pre-projected bf16 Q,K,V in [B, NH, S, DH] layout.
// Q is pre-scaled by 1/sqrt(DH). Each wave owns 16 q-rows; block of 4 waves
// covers 64 q-rows of one (b,h). Online softmax, key tiles of 32.
// V tiles are staged into per-wave LDS with GLOBAL_LOAD_ASYNC_TO_LDS_B128,
// issued before the score WMMAs and waited (ASYNCcnt) only before P@V.
// Output ctx: bf16 [B, S, NH, DH] (== [M, HID] row-major for final GEMM).
// ---------------------------------------------------------------------------
__global__ __launch_bounds__(128)
void attention_wmma_kernel(const u16* __restrict__ Q, const u16* __restrict__ K,
                           const u16* __restrict__ V, u16* __restrict__ ctx,
                           const int* __restrict__ seq_len,
                           const int* __restrict__ lex_num) {
    __shared__ __align__(16) float pbuf[4][16][32];   // P tile per wave (8 KB)
    __shared__ __align__(16) u16   vbuf[4][32][64];   // V tile per wave (16 KB)

    const int lane = threadIdx.x & 31;
    const int wv   = threadIdx.x >> 5;
    const int col  = lane & 15;
    const int hi16 = lane >> 4;

    const int qblocks = S_ / 64;
    const int blk = blockIdx.x;
    const int qb  = blk % qblocks;
    const int bh  = blk / qblocks;
    const int h   = bh % NH_;
    const int b   = bh / NH_;

    const size_t base = (size_t)(b * NH_ + h) * S_ * DH_;
    const int q0 = qb * 64 + wv * 16;
    const int limit = seq_len[b] + lex_num[0];

    // Load the wave's Q tile as two 16x32 A fragments (d = [0,32) and [32,64))
    v16bf qa[2];
    {
        const u16* qrow = Q + base + (size_t)(q0 + col) * DH_;
#pragma unroll
        for (int c = 0; c < 2; ++c) {
            uint4 lo = *(const uint4*)(qrow + c * 32 + hi16 * 8);
            uint4 hi = *(const uint4*)(qrow + c * 32 + 16 + hi16 * 8);
            qa[c] = mkfrag(lo, hi);
        }
    }

    v8f zero = {0.f,0.f,0.f,0.f,0.f,0.f,0.f,0.f};
    v8f o[4] = {zero, zero, zero, zero};            // 16 x 64 output accumulator
    float mrow[8], lrow[8];
#pragma unroll
    for (int r = 0; r < 8; ++r) { mrow[r] = -3.0e38f; lrow[r] = 0.f; }

    for (int kt = 0; kt < S_; kt += 32) {
        // Stage V tile (32 keys x 64 d): lane owns one 128B row.
        {
            const u16* src = V + base + (size_t)(kt + lane) * DH_;
            u16* dstl = &vbuf[wv][lane][0];
#if HAVE_ASYNC
#pragma unroll
            for (int i = 0; i < 8; ++i)
                __builtin_amdgcn_global_load_async_to_lds_b128(
                    (AS1 i32x4*)(src + i * 8), (AS3 i32x4*)(dstl + i * 8), 0, 0);
#else
#pragma unroll
            for (int i = 0; i < 8; ++i)
                ((uint4*)dstl)[i] = ((const uint4*)src)[i];
#endif
        }

        // Scores: 16 q-rows x 32 key-cols, two 16-col WMMA groups, d reduced in 2 chunks
        v8f sc[2];
#pragma unroll
        for (int g = 0; g < 2; ++g) {
            const u16* krow = K + base + (size_t)(kt + g * 16 + col) * DH_;
            v8f acc = zero;
#pragma unroll
            for (int c = 0; c < 2; ++c) {
                uint4 lo = *(const uint4*)(krow + c * 32 + hi16 * 8);
                uint4 hi = *(const uint4*)(krow + c * 32 + 16 + hi16 * 8);
                acc = wmma_bf16(qa[c], mkfrag(lo, hi), acc);
            }
            sc[g] = acc;
        }

        // Padding mask on key axis (one column per lane per group)
#pragma unroll
        for (int g = 0; g < 2; ++g) {
            if (kt + g * 16 + col >= limit) {
#pragma unroll
                for (int r = 0; r < 8; ++r) sc[g][r] = -1.0e30f;
            }
        }

        // Online softmax, one reduction per owned row (8 rows/lane, 16-lane halves)
#pragma unroll
        for (int r = 0; r < 8; ++r) {
            float mx = fmaxf(sc[0][r], sc[1][r]);
#pragma unroll
            for (int msk = 8; msk >= 1; msk >>= 1)
                mx = fmaxf(mx, __shfl_xor(mx, msk, 16));
            const float mnew = fmaxf(mrow[r], mx);
            const float corr = __expf(mrow[r] - mnew);
            const float p0 = __expf(sc[0][r] - mnew);
            const float p1 = __expf(sc[1][r] - mnew);
            float s = p0 + p1;
#pragma unroll
            for (int msk = 8; msk >= 1; msk >>= 1)
                s += __shfl_xor(s, msk, 16);
            lrow[r] = lrow[r] * corr + s;
            mrow[r] = mnew;
            sc[0][r] = p0; sc[1][r] = p1;
#pragma unroll
            for (int c = 0; c < 4; ++c) o[c][r] *= corr;
        }

        // Re-layout P: C-fragment (f32) -> LDS -> A-fragment (bf16)
#pragma unroll
        for (int g = 0; g < 2; ++g)
#pragma unroll
            for (int r = 0; r < 8; ++r)
                pbuf[wv][r + 8 * hi16][g * 16 + col] = sc[g][r];

        asm volatile("s_wait_dscnt 0" ::: "memory");   // in-wave LDS RAW (P tile)

        v16bf pa;
        {
            v16us pu;
#pragma unroll
            for (int e = 0; e < 16; ++e) {
                const int kk = hi16 * 16 + e;
                pu[e] = f2bf(pbuf[wv][col][kk]);
            }
            pa = __builtin_bit_cast(v16bf, pu);
        }

#if HAVE_ASYNC
        wait_async0();   // V tile DMA has been hiding under QK^T + softmax
#endif

        // o += P @ V : four 16-d-col groups, B fragments gathered from LDS
#pragma unroll
        for (int c = 0; c < 4; ++c) {
            v16us vu;
#pragma unroll
            for (int e = 0; e < 16; ++e)
                vu[e] = vbuf[wv][hi16 * 16 + e][c * 16 + col];
            o[c] = wmma_bf16(pa, __builtin_bit_cast(v16bf, vu), o[c]);
        }
    }

    // Normalize and store context as bf16 [B, S, NH, DH]
#pragma unroll
    for (int r = 0; r < 8; ++r) {
        const float inv = 1.0f / lrow[r];
        const int row = q0 + r + 8 * hi16;
        u16* dst = ctx + ((size_t)(b * S_ + row) * NH_ + h) * DH_;
#pragma unroll
        for (int c = 0; c < 4; ++c)
            dst[c * 16 + col] = f2bf(o[c][r] * inv);
    }
}

// ---------------------------------------------------------------------------
// Host launcher
// ---------------------------------------------------------------------------
extern "C" void kernel_launch(void* const* d_in, const int* in_sizes, int n_in,
                              void* d_out, int out_size, void* d_ws, size_t ws_size,
                              hipStream_t stream) {
    const float* key   = (const float*)d_in[0];
    const float* query = (const float*)d_in[1];
    const float* value = (const float*)d_in[2];
    const float* Wk = (const float*)d_in[3];
    const float* bk = (const float*)d_in[4];
    const float* Wq = (const float*)d_in[5];
    const float* bq = (const float*)d_in[6];
    const float* Wv = (const float*)d_in[7];
    const float* bv = (const float*)d_in[8];
    const float* Wo = (const float*)d_in[9];
    const float* bo = (const float*)d_in[10];
    const int* seq_len = (const int*)d_in[11];
    const int* lex_num = (const int*)d_in[12];
    float* out = (float*)d_out;

    const size_t MH = (size_t)M_ * HID_;     // 8M elems
    const size_t WH = (size_t)HID_ * HID_;   // 1M elems

    char* ws = (char*)d_ws;
    u16* keyb = (u16*)ws;  ws += MH * 2;
    u16* qryb = (u16*)ws;  ws += MH * 2;
    u16* valb = (u16*)ws;  ws += MH * 2;
    u16* Wkb  = (u16*)ws;  ws += WH * 2;
    u16* Wqb  = (u16*)ws;  ws += WH * 2;
    u16* Wvb  = (u16*)ws;  ws += WH * 2;
    u16* Wob  = (u16*)ws;  ws += WH * 2;
    u16* Qp   = (u16*)ws;  ws += MH * 2;
    u16* Kp   = (u16*)ws;  ws += MH * 2;
    u16* Vp   = (u16*)ws;  ws += MH * 2;
    u16* ctx  = keyb;  // keyb's last reader is the K projection; attention (ctx
                       // writer) runs strictly after all projections.

    // 1) fp32 -> bf16 conversions
    {
        const int thr = 256;
        const int gA = (int)((MH / 4 + thr - 1) / thr);
        const int gW = (int)((WH / 4 + thr - 1) / thr);
        f32_to_bf16_kernel<<<gA, thr, 0, stream>>>(key,   keyb, (int)MH);
        f32_to_bf16_kernel<<<gA, thr, 0, stream>>>(query, qryb, (int)MH);
        f32_to_bf16_kernel<<<gA, thr, 0, stream>>>(value, valb, (int)MH);
        f32_to_bf16_kernel<<<gW, thr, 0, stream>>>(Wk, Wkb, (int)WH);
        f32_to_bf16_kernel<<<gW, thr, 0, stream>>>(Wq, Wqb, (int)WH);
        f32_to_bf16_kernel<<<gW, thr, 0, stream>>>(Wv, Wvb, (int)WH);
        f32_to_bf16_kernel<<<gW, thr, 0, stream>>>(Wo, Wob, (int)WH);
    }

    // 2) Q/K/V projections (store bf16 [B,NH,S,DH]; Q pre-scaled by 1/sqrt(64))
    {
        dim3 grid(HID_ / 64, M_ / 64);
        dim3 blockd(128);
        linear_wmma_kernel<<<grid, blockd, 0, stream>>>(keyb, Wkb, bk, nullptr, Kp,
                                                        M_, HID_, HID_, 1.0f, 1);
        linear_wmma_kernel<<<grid, blockd, 0, stream>>>(qryb, Wqb, bq, nullptr, Qp,
                                                        M_, HID_, HID_, 0.125f, 1);
        linear_wmma_kernel<<<grid, blockd, 0, stream>>>(valb, Wvb, bv, nullptr, Vp,
                                                        M_, HID_, HID_, 1.0f, 1);
    }

    // 3) Flash attention -> ctx (bf16 [B,S,NH,DH] == [M, HID])
    attention_wmma_kernel<<<dim3(B_ * NH_ * (S_ / 64)), dim3(128), 0, stream>>>(
        Qp, Kp, Vp, ctx, seq_len, lex_num);

    // 4) Output projection -> fp32 d_out
    {
        dim3 grid(HID_ / 64, M_ / 64);
        dim3 blockd(128);
        linear_wmma_kernel<<<grid, blockd, 0, stream>>>(ctx, Wob, bo, out, nullptr,
                                                        M_, HID_, HID_, 1.0f, 0);
    }
    (void)in_sizes; (void)n_in; (void)out_size; (void)ws_size;
}
